// EmformerDistillModel_80221399155292
// MI455X (gfx1250) — matrix operand order repaired and checked by reading, not compiled, not run on hardware
//
#include <hip/hip_runtime.h>

// ---------------- model constants ----------------
#define B_      4
#define U_      512
#define D_      80
#define H_      8
#define FFN_    2048
#define L_      12
#define SEG_    32
#define LC_     50
#define RC_     8
#define NSEG_   16
#define RCLEN_  128          // RC * NSEG
#define KSEQ_   640          // RCLEN + U
#define OUTD_   768
#define DPAD_   96           // D padded to multiple of 32
#define QKV_    240          // q(80) + k(80) + v(80)
#define EPS_    1e-5f
#define SCALE_  0.31622776601683794f   // (D/H)^-0.5 = 10^-0.5
#define BIGM_   (1 << 30)

typedef __attribute__((ext_vector_type(16))) __bf16 v16bf;
typedef __attribute__((ext_vector_type(8)))  float  v8f;

struct u128 { unsigned int x, y, z, w; };

__device__ inline v16bf pack_bf32(u128 a, u128 b) {
  union { u128 q[2]; v16bf v; } u;
  u.q[0] = a; u.q[1] = b;
  return u.v;
}

#define V8F_ZERO {0.f, 0.f, 0.f, 0.f, 0.f, 0.f, 0.f, 0.f}

// ---------------- weight convert f32 -> bf16 with K zero-pad ----------------
__global__ __launch_bounds__(256) void wconv_bf16(
    const float* __restrict__ src, __bf16* __restrict__ dst,
    int rows, int kin, int kpad) {
  long idx = (long)blockIdx.x * 256 + threadIdx.x;
  long total = (long)rows * kpad;
  if (idx >= total) return;
  int  k = (int)(idx % kpad);
  long r = idx / kpad;
  dst[idx] = (k < kin) ? (__bf16)src[r * (long)kin + k] : (__bf16)0.f;
}

// ---------------- build x = concat(rc, utt) from mel_input ----------------
__global__ __launch_bounds__(256) void build_x(
    const float* __restrict__ mel, float* __restrict__ x) {
  int idx = blockIdx.x * 256 + threadIdx.x;           // over B*640*80
  if (idx >= B_ * KSEQ_ * D_) return;
  int d   = idx % D_;
  int row = (idx / D_) % KSEQ_;
  int b   = idx / (KSEQ_ * D_);
  int src;
  if (row < RCLEN_) {                                  // right-context rows
    int i = row >> 3, r = row & 7;
    src = (i < NSEG_ - 1) ? (i + 1) * SEG_ + r : U_ + r;   // last seg: T-RC..T-1
  } else {
    src = row - RCLEN_;
  }
  x[idx] = mel[((long)b * (U_ + RC_) + src) * D_ + d];
}

// ---------------- LayerNorm over D=80, wave32 per row ----------------
__global__ __launch_bounds__(256) void ln_rows(
    const float* __restrict__ in, const float* __restrict__ g,
    const float* __restrict__ b, float* __restrict__ outF,
    __bf16* __restrict__ outB, int nrows) {
  int wave = threadIdx.x >> 5;
  int l    = threadIdx.x & 31;
  int row  = blockIdx.x * 8 + wave;
  if (row >= nrows) return;
  const float* xr = in + (long)row * D_;
  float e0 = xr[l];
  float e1 = xr[32 + l];
  float e2 = (l < 16) ? xr[64 + l] : 0.f;
  float s = e0 + e1 + e2;
#pragma unroll
  for (int off = 16; off > 0; off >>= 1) s += __shfl_xor(s, off);
  float mean = s * (1.f / D_);
  float d0 = e0 - mean, d1 = e1 - mean, d2 = (l < 16) ? (e2 - mean) : 0.f;
  float v = d0 * d0 + d1 * d1 + d2 * d2;
#pragma unroll
  for (int off = 16; off > 0; off >>= 1) v += __shfl_xor(v, off);
  float rstd = rsqrtf(v * (1.f / D_) + EPS_);

  float o0 = d0 * rstd * g[l]      + b[l];
  float o1 = d1 * rstd * g[32 + l] + b[32 + l];
  if (outF) {
    outF[(long)row * D_ + l]      = o0;
    outF[(long)row * D_ + 32 + l] = o1;
  }
  if (outB) {
    outB[(long)row * DPAD_ + l]      = (__bf16)o0;
    outB[(long)row * DPAD_ + 32 + l] = (__bf16)o1;
  }
  if (l < 16) {
    float o2 = d2 * rstd * g[64 + l] + b[64 + l];
    if (outF) outF[(long)row * D_ + 64 + l] = o2;
    if (outB) outB[(long)row * DPAD_ + 64 + l] = (__bf16)o2;
  } else {
    if (outB) outB[(long)row * DPAD_ + 64 + l] = (__bf16)0.f;  // cols 80..95
  }
}

// ---------------- multi-tile WMMA bf16 GEMM: out = A @ W^T (+bias)(+res)(relu)
// MT x NT tiles of 16x16 per wave; one operand side is reused across tiles.
// Double-buffered K loop: next K-tile operands load while current WMMAs run.
//   A operand: lane = row (l&15); halves K = half*8 + {0..7, 16..23}  -> 2x b128
//   B operand: lane = col (l&15); halves K = half*8 + {0..7, 16..23}  -> 2x b128
template <int MT, int NT>
__global__ __launch_bounds__(32) void gemm_wmma_bf16(
    const __bf16* __restrict__ A, int lda,
    const __bf16* __restrict__ W, int ldw,
    const float* __restrict__ bias,
    const float* __restrict__ resid, int ldres,
    float* __restrict__ outF, int ldoF,
    __bf16* __restrict__ outB, int ldoB,
    int kTiles, int doRelu, int n0,
    int m_batch, int a_row_start, int a_batch_rows) {
  int mt0 = blockIdx.x * MT, nt0 = blockIdx.y * NT;
  int l = threadIdx.x, half = l >> 4, lr = l & 15;

  const __bf16* Ar[MT];
  const __bf16* Wr[NT];
#pragma unroll
  for (int i = 0; i < MT; ++i) {
    int  gr_l = (mt0 + i) * 16 + lr;
    long arow = (long)(gr_l / m_batch) * a_batch_rows + a_row_start + (gr_l % m_batch);
    Ar[i] = A + arow * (long)lda + half * 8;
  }
#pragma unroll
  for (int j = 0; j < NT; ++j)
    Wr[j] = W + (long)((nt0 + j) * 16 + lr) * ldw + half * 8;

  const v8f vzero = V8F_ZERO;
  v8f acc[MT][NT];
#pragma unroll
  for (int i = 0; i < MT; ++i)
#pragma unroll
    for (int j = 0; j < NT; ++j) acc[i][j] = vzero;

  // prologue: K-tile 0
  u128 a0[MT], a1[MT], b0[NT], b1[NT];
#pragma unroll
  for (int i = 0; i < MT; ++i) {
    a0[i] = *(const u128*)(Ar[i]);
    a1[i] = *(const u128*)(Ar[i] + 16);
  }
#pragma unroll
  for (int j = 0; j < NT; ++j) {
    b0[j] = *(const u128*)(Wr[j]);
    b1[j] = *(const u128*)(Wr[j] + 16);
  }

  for (int kt = 0; kt < kTiles; ++kt) {
    u128 na0[MT], na1[MT], nb0[NT], nb1[NT];
    if (kt + 1 < kTiles) {                 // issue next tile's loads first
      int ko = (kt + 1) * 32;
#pragma unroll
      for (int i = 0; i < MT; ++i) {
        na0[i] = *(const u128*)(Ar[i] + ko);
        na1[i] = *(const u128*)(Ar[i] + ko + 16);
      }
#pragma unroll
      for (int j = 0; j < NT; ++j) {
        nb0[j] = *(const u128*)(Wr[j] + ko);
        nb1[j] = *(const u128*)(Wr[j] + ko + 16);
      }
      if (kt + 2 < kTiles) {
#pragma unroll
        for (int j = 0; j < NT; ++j)
          __builtin_prefetch(Wr[j] + (kt + 2) * 32, 0, 3);   // near-cache prefetch
      }
    }
    v16bf av[MT], bv[NT];
#pragma unroll
    for (int i = 0; i < MT; ++i) av[i] = pack_bf32(a0[i], a1[i]);
#pragma unroll
    for (int j = 0; j < NT; ++j) bv[j] = pack_bf32(b0[j], b1[j]);
#pragma unroll
    for (int i = 0; i < MT; ++i)
#pragma unroll
      for (int j = 0; j < NT; ++j)
        acc[i][j] = __builtin_amdgcn_wmma_f32_16x16x32_bf16(
            false, av[i], false, bv[j], (short)0, acc[i][j], false, false);
#pragma unroll
    for (int i = 0; i < MT; ++i) { a0[i] = na0[i]; a1[i] = na1[i]; }
#pragma unroll
    for (int j = 0; j < NT; ++j) { b0[j] = nb0[j]; b1[j] = nb1[j]; }
  }

  // epilogue
#pragma unroll
  for (int j = 0; j < NT; ++j) {
    int nloc = (nt0 + j) * 16 + lr;
    int col  = n0 + nloc;
    float bvv = bias ? bias[nloc] : 0.f;
#pragma unroll
    for (int i = 0; i < MT; ++i) {
#pragma unroll
      for (int gg = 0; gg < 8; ++gg) {
        int m  = gg + half * 8;
        int gr = (mt0 + i) * 16 + m;
        float v = acc[i][j][gg] + bvv;
        if (resid)  v += resid[(long)gr * ldres + col];
        if (doRelu && v < 0.f) v = 0.f;
        if (outF) outF[(long)gr * ldoF + col] = v;
        if (outB) outB[(long)gr * ldoB + col] = (__bf16)v;
      }
    }
  }
}

// ---------------- repack qkv -> padded per-head Q,K and transposed V^T -------
__global__ __launch_bounds__(256) void repack_qkv(
    const __bf16* __restrict__ qkv, __bf16* __restrict__ qp,
    __bf16* __restrict__ kp, __bf16* __restrict__ vT) {
  int idx = blockIdx.x * 256 + threadIdx.x;     // B*640*H*32
  if (idx >= B_ * KSEQ_ * H_ * 32) return;
  int d = idx & 31;
  int h = (idx >> 5) & 7;
  int t = (idx >> 8) % KSEQ_;
  int b = (idx >> 8) / KSEQ_;
  long src = ((long)b * KSEQ_ + t) * QKV_;
  long bh  = (long)(b * H_ + h);
  if (d < D_ / H_) {
    float q = (float)qkv[src + h * (D_ / H_) + d];
    qp[(bh * KSEQ_ + t) * 32 + d] = (__bf16)(q * SCALE_);
    kp[(bh * KSEQ_ + t) * 32 + d] = qkv[src + 80 + h * (D_ / H_) + d];
  } else {
    qp[(bh * KSEQ_ + t) * 32 + d] = (__bf16)0.f;
    kp[(bh * KSEQ_ + t) * 32 + d] = (__bf16)0.f;
  }
  if (d < 16) {
    __bf16 vv = (d < D_ / H_) ? qkv[src + 160 + h * (D_ / H_) + d] : (__bf16)0.f;
    vT[(bh * 16 + d) * KSEQ_ + t] = vv;
  }
}

// ---------------- static Emformer attention mask (allowed = !mask) -----------
__device__ inline bool attn_allowed(int row, int col) {
  int seg = (row < RCLEN_) ? (row >> 3) : ((row - RCLEN_) >> 5);
  if (col < RCLEN_) return (col >> 3) == seg;
  int u = col - RCLEN_;
  int s = seg * SEG_ - LC_; if (s < 0) s = 0;
  int e = (seg + 1) * SEG_; if (e > U_) e = U_;
  return (u >= s) && (u < e);
}

// ---------------- fused attention: scores -> mask -> softmax -> P@V ----------
__global__ __launch_bounds__(32) void attn_wmma(
    const __bf16* __restrict__ qp, const __bf16* __restrict__ kp,
    const __bf16* __restrict__ vT, const int* __restrict__ lengths,
    __bf16* __restrict__ attnOut) {
  __shared__ float  sS[16 * KSEQ_];     // 40 KB
  __shared__ __bf16 sP[16 * KSEQ_];     // 20 KB
  __shared__ float  sRinv[16];

  int qt = blockIdx.x, h = blockIdx.y, b = blockIdx.z;
  int l = threadIdx.x, half = l >> 4, lr = l & 15;
  long bh = (long)(b * H_ + h);
  const __bf16* Q  = qp + (bh * KSEQ_) * 32;
  const __bf16* Kk = kp + (bh * KSEQ_) * 32;
  const __bf16* V  = vT + (bh * 16) * KSEQ_;
  int kvalid = lengths[b] + RCLEN_;

  // A operand (16x32, K = padded head dim): loaded once.
  const __bf16* Qr = Q + (long)(qt * 16 + lr) * 32;
  v16bf aq = pack_bf32(*(const u128*)(Qr + half * 8),
                       *(const u128*)(Qr + half * 8 + 16));

  const v8f vzero = V8F_ZERO;

  // ---- scores: 40 tiles of 16 keys, masked into LDS ----
  for (int kt = 0; kt < 40; ++kt) {
    const __bf16* Kr = Kk + (long)(kt * 16 + lr) * 32;
    v16bf bk = pack_bf32(*(const u128*)(Kr + half * 8),
                         *(const u128*)(Kr + half * 8 + 16));
    v8f c = vzero;
    c = __builtin_amdgcn_wmma_f32_16x16x32_bf16(false, aq, false, bk,
                                                (short)0, c, false, false);
#pragma unroll
    for (int gg = 0; gg < 8; ++gg) {
      int m   = gg + half * 8;
      int row = qt * 16 + m;
      int col = kt * 16 + lr;
      float sv = c[gg];
      if (!attn_allowed(row, col)) sv = -__builtin_huge_valf();
      if (col >= kvalid)           sv = -1e8f;      // pad overrides -inf (ref order)
      sS[m * KSEQ_ + col] = sv;
    }
  }
  __syncthreads();

  // ---- softmax: row = lr, each half-wave covers 320 columns ----
  {
    const float* rowp = sS + lr * KSEQ_ + half * 320;
    float mx = -__builtin_huge_valf();
    for (int c = 0; c < 320; ++c) mx = fmaxf(mx, rowp[c]);
    mx = fmaxf(mx, __shfl_xor(mx, 16));
    float sum = 0.f;
    __bf16* prow = sP + lr * KSEQ_ + half * 320;
    for (int c = 0; c < 320; ++c) {
      float e = expf(rowp[c] - mx);
      sum += e;
      prow[c] = (__bf16)e;
    }
    sum += __shfl_xor(sum, 16);
    if (half == 0) sRinv[lr] = 1.f / fmaxf(sum, 1e-20f);
  }
  __syncthreads();

  // ---- P @ V: 20 K-tiles of 32 keys; two accumulators break the RAW chain ---
  v8f acc0 = vzero, acc1 = vzero;
  for (int kt = 0; kt < 20; kt += 2) {
    int ka0 = kt * 32 + half * 8;
    v16bf ap0 = pack_bf32(*(const u128*)(sP + lr * KSEQ_ + ka0),
                          *(const u128*)(sP + lr * KSEQ_ + ka0 + 16));
    const __bf16* Vr0 = V + (long)lr * KSEQ_ + kt * 32 + half * 8;
    v16bf bv0 = pack_bf32(*(const u128*)(Vr0), *(const u128*)(Vr0 + 16));
    int ka1 = (kt + 1) * 32 + half * 8;
    v16bf ap1 = pack_bf32(*(const u128*)(sP + lr * KSEQ_ + ka1),
                          *(const u128*)(sP + lr * KSEQ_ + ka1 + 16));
    const __bf16* Vr1 = V + (long)lr * KSEQ_ + (kt + 1) * 32 + half * 8;
    v16bf bv1 = pack_bf32(*(const u128*)(Vr1), *(const u128*)(Vr1 + 16));
    acc0 = __builtin_amdgcn_wmma_f32_16x16x32_bf16(false, ap0, false, bv0,
                                                   (short)0, acc0, false, false);
    acc1 = __builtin_amdgcn_wmma_f32_16x16x32_bf16(false, ap1, false, bv1,
                                                   (short)0, acc1, false, false);
  }
  v8f accv = acc0 + acc1;

  // ---- write attn output (bf16, ld 96), valid head dims only ----
  if (lr < D_ / H_) {
#pragma unroll
    for (int gg = 0; gg < 8; ++gg) {
      int m = gg + half * 8;
      attnOut[((long)b * KSEQ_ + qt * 16 + m) * DPAD_ + h * (D_ / H_) + lr] =
          (__bf16)(accv[gg] * sRinv[m]);
    }
  }
}

// ---------------- append lengths to d_out ----------------
__global__ void tail_lengths(const int* __restrict__ len, int* __restrict__ dst) {
  if (threadIdx.x < B_) dst[threadIdx.x] = len[threadIdx.x];
}

// =====================================================================
extern "C" void kernel_launch(void* const* d_in, const int* in_sizes, int n_in,
                              void* d_out, int out_size, void* d_ws, size_t ws_size,
                              hipStream_t stream) {
  (void)in_sizes; (void)n_in; (void)out_size; (void)ws_size;
  const float* mel    = (const float*)d_in[0];
  const float* ln_in_g = (const float*)d_in[1];
  const float* ln_in_b = (const float*)d_in[2];
  const float* Wq  = (const float*)d_in[3];
  const float* bq  = (const float*)d_in[4];
  const float* Wkv = (const float*)d_in[5];
  const float* bkv = (const float*)d_in[6];
  const float* Wo  = (const float*)d_in[7];
  const float* bo  = (const float*)d_in[8];
  const float* ln_ff_g = (const float*)d_in[9];
  const float* ln_ff_b = (const float*)d_in[10];
  const float* W1 = (const float*)d_in[11];
  const float* b1 = (const float*)d_in[12];
  const float* W2 = (const float*)d_in[13];
  const float* b2 = (const float*)d_in[14];
  const float* ln_out_g = (const float*)d_in[15];
  const float* ln_out_b = (const float*)d_in[16];
  const float* Wp = (const float*)d_in[17];
  const float* bp = (const float*)d_in[18];
  const int*   lengths = (const int*)d_in[19];

  // ---- workspace carve (256B aligned) ----
  char* p = (char*)d_ws;
  auto alloc = [&](size_t bytes) -> void* {
    void* r = (void*)p;
    p += (bytes + 255) & ~(size_t)255;
    return r;
  };
  const int M = B_ * KSEQ_;                      // 2560 rows
  __bf16* wq_bf  = (__bf16*)alloc((size_t)L_ * D_   * DPAD_ * 2);
  __bf16* wkv_bf = (__bf16*)alloc((size_t)L_ * 2*D_ * DPAD_ * 2);
  __bf16* wo_bf  = (__bf16*)alloc((size_t)L_ * D_   * DPAD_ * 2);
  __bf16* w1_bf  = (__bf16*)alloc((size_t)L_ * FFN_ * DPAD_ * 2);
  __bf16* w2_bf  = (__bf16*)alloc((size_t)L_ * D_   * FFN_  * 2);
  __bf16* wp_bf  = (__bf16*)alloc((size_t)OUTD_ * DPAD_ * 2);
  float*  x      = (float*) alloc((size_t)M * D_ * 4);
  float*  y      = (float*) alloc((size_t)M * D_ * 4);
  __bf16* lnbuf  = (__bf16*)alloc((size_t)M * DPAD_ * 2);
  __bf16* lnobf  = (__bf16*)alloc((size_t)M * DPAD_ * 2);
  __bf16* qkv    = (__bf16*)alloc((size_t)M * QKV_ * 2);
  __bf16* qpad   = (__bf16*)alloc((size_t)B_ * H_ * KSEQ_ * 32 * 2);
  __bf16* kpad   = (__bf16*)alloc((size_t)B_ * H_ * KSEQ_ * 32 * 2);
  __bf16* vT     = (__bf16*)alloc((size_t)B_ * H_ * 16 * KSEQ_ * 2);
  __bf16* attnb  = (__bf16*)alloc((size_t)M * DPAD_ * 2);
  __bf16* hbuf   = (__bf16*)alloc((size_t)M * FFN_ * 2);

  auto cdiv = [](long a, long b) { return (int)((a + b - 1) / b); };

  // ---- one-shot bf16 weight conversion (per launch; deterministic) ----
  wconv_bf16<<<cdiv((long)L_*D_*DPAD_, 256), 256, 0, stream>>>(Wq,  wq_bf,  L_*D_,   D_,   DPAD_);
  wconv_bf16<<<cdiv((long)L_*2*D_*DPAD_,256),256, 0, stream>>>(Wkv, wkv_bf, L_*2*D_, D_,   DPAD_);
  wconv_bf16<<<cdiv((long)L_*D_*DPAD_, 256), 256, 0, stream>>>(Wo,  wo_bf,  L_*D_,   D_,   DPAD_);
  wconv_bf16<<<cdiv((long)L_*FFN_*DPAD_,256),256, 0, stream>>>(W1,  w1_bf,  L_*FFN_, D_,   DPAD_);
  wconv_bf16<<<cdiv((long)L_*D_*FFN_, 256), 256, 0, stream>>>(W2,  w2_bf,  L_*D_,   FFN_, FFN_);
  wconv_bf16<<<cdiv((long)OUTD_*DPAD_,256), 256, 0, stream>>>(Wp,  wp_bf,  OUTD_,   D_,   DPAD_);

  build_x<<<cdiv((long)M * D_, 256), 256, 0, stream>>>(mel, x);
  hipMemsetAsync(attnb, 0, (size_t)M * DPAD_ * 2, stream);   // zero pad cols once

  // MT=4: 64 rows x 16 cols per wave (weights reused) — for N in {80,160}
  auto gemmM4 = [&](const __bf16* A, int lda, const __bf16* W, int ldw,
                    const float* bias, const float* resid, int ldres,
                    float* outF, int ldoF, __bf16* outB, int ldoB,
                    int kTiles, int relu, int n0, int Mtiles, int Ntiles) {
    gemm_wmma_bf16<4, 1><<<dim3(Mtiles / 4, Ntiles), 32, 0, stream>>>(
        A, lda, W, ldw, bias, resid, ldres, outF, ldoF, outB, ldoB,
        kTiles, relu, n0, BIGM_, 0, M);
  };
  // NT=4: 16 rows x 64 cols per wave (activations reused) — for N mult of 64
  auto gemmN4 = [&](const __bf16* A, int lda, const __bf16* W, int ldw,
                    const float* bias, const float* resid, int ldres,
                    float* outF, int ldoF, __bf16* outB, int ldoB,
                    int kTiles, int relu, int n0, int Mtiles, int Ntiles,
                    int m_batch, int a_row_start, int a_batch_rows) {
    gemm_wmma_bf16<1, 4><<<dim3(Mtiles, Ntiles / 4), 32, 0, stream>>>(
        A, lda, W, ldw, bias, resid, ldres, outF, ldoF, outB, ldoB,
        kTiles, relu, n0, m_batch, a_row_start, a_batch_rows);
  };

  const int Mt = M / 16;           // 160
  for (int l = 0; l < L_; ++l) {
    // 1) LN-in -> bf16 (pad 96)
    ln_rows<<<M / 8, 256, 0, stream>>>(x, ln_in_g + l*D_, ln_in_b + l*D_,
                                       nullptr, lnbuf, M);
    // 2) Q projection (cols 0..79 of qkv), 3) KV projection (cols 80..239)
    gemmM4(lnbuf, DPAD_, wq_bf  + (long)l*D_*DPAD_,   DPAD_, bq  + l*D_,
           nullptr, 0, nullptr, 0, qkv, QKV_, 3, 0, 0,  Mt, D_/16);
    gemmM4(lnbuf, DPAD_, wkv_bf + (long)l*2*D_*DPAD_, DPAD_, bkv + l*2*D_,
           nullptr, 0, nullptr, 0, qkv, QKV_, 3, 0, 80, Mt, 2*D_/16);
    // 4) repack into WMMA-friendly padded head layouts
    repack_qkv<<<cdiv((long)B_*KSEQ_*H_*32, 256), 256, 0, stream>>>(qkv, qpad, kpad, vT);
    // 5) fused masked attention
    attn_wmma<<<dim3(KSEQ_/16, H_, B_), 32, 0, stream>>>(qpad, kpad, vT, lengths, attnb);
    // 6) output projection + residual -> y (f32)
    gemmM4(attnb, DPAD_, wo_bf + (long)l*D_*DPAD_, DPAD_, bo + l*D_,
           x, D_, y, D_, nullptr, 0, 3, 0, 0, Mt, D_/16);
    // 7) LN-ff -> bf16
    ln_rows<<<M / 8, 256, 0, stream>>>(y, ln_ff_g + l*D_, ln_ff_b + l*D_,
                                       nullptr, lnbuf, M);
    // 8) FFN up + ReLU -> bf16 hbuf (N=2048, activations reused across 4 tiles)
    gemmN4(lnbuf, DPAD_, w1_bf + (long)l*FFN_*DPAD_, DPAD_, b1 + l*FFN_,
           nullptr, 0, nullptr, 0, hbuf, FFN_, 3, 1, 0, Mt, FFN_/16,
           BIGM_, 0, M);
    // 9) FFN down + residual -> x (f32)  (K=2048 -> 64 K-tiles)
    gemmM4(hbuf, FFN_, w2_bf + (long)l*D_*FFN_, FFN_, b2 + l*D_,
           y, D_, x, D_, nullptr, 0, FFN_/32, 0, 0, Mt, D_/16);
    // 10) LN-out: f32 carry (in place) + bf16 copy for final projection
    ln_rows<<<M / 8, 256, 0, stream>>>(x, ln_out_g + l*D_, ln_out_b + l*D_,
                                       x, lnobf, M);
  }

  // ---- final projection: utt rows (128..639 per batch) @ Wp^T + bp ----
  gemmN4(lnobf, DPAD_, wp_bf, DPAD_, bp, nullptr, 0,
         (float*)d_out, OUTD_, nullptr, 0, 3, 0, 0,
         (B_ * U_) / 16, OUTD_ / 16, U_, RCLEN_, KSEQ_);

  // ---- append lengths (int32 bit-pattern) after the f32 output ----
  tail_lengths<<<1, 32, 0, stream>>>(lengths, (int*)d_out + (long)B_ * U_ * OUTD_);
}
